// GroveMoeSparseMoeBlock_46127948759731
// MI455X (gfx1250) — matrix operand order repaired and checked by reading, not compiled, not run on hardware
//
#include <hip/hip_runtime.h>
#include <hip/hip_bf16.h>
#include <math.h>

// ---------------- problem constants ----------------
constexpr int kT  = 4096;   // tokens (B*S)
constexpr int kH  = 2048;   // hidden
constexpr int kE  = 32;     // large experts
constexpr int kI  = 768;    // large intermediate
constexpr int kNC = 16;     // chunk experts
constexpr int kCI = 128;    // chunk intermediate
constexpr int kTOPK = 4;
constexpr float kSMALL = 0.05f;

typedef __bf16 bf16_t;
typedef __attribute__((ext_vector_type(16))) __bf16 v16bf;
typedef __attribute__((ext_vector_type(8)))  float  v8f;

// ---------------- helpers ----------------
__device__ inline unsigned short f2bfbits(float f){
  unsigned u; __builtin_memcpy(&u,&f,4);
  unsigned r = u + 0x7FFFu + ((u>>16)&1u);   // round-to-nearest-even
  return (unsigned short)(r>>16);
}
__device__ inline unsigned pack2bf(float a, float b){
  return (unsigned)f2bfbits(a) | ((unsigned)f2bfbits(b) << 16);
}
__device__ inline float sigmoidf_(float x){ return 1.f/(1.f+__expf(-x)); }
__device__ inline float siluf_(float x){ return x * sigmoidf_(x); }

__device__ inline v8f wmma_bf16(v16bf a, v16bf b, v8f c){
  return __builtin_amdgcn_wmma_f32_16x16x32_bf16(false, a, false, b, (short)0, c, false, false);
}

// A fragment (16x32 bf16) loaded straight from a row-major bf16 row in global memory.
// Lane layout: lane m=lane&15; lo lanes hold K 0..15, hi lanes K 16..31 (matches packed B).
// Both 8-element runs are contiguous 16B -> two global_load_b128 per lane.
__device__ inline v16bf load_a_frag_g(const unsigned short* __restrict__ rowp, int kbase, int hi){
  v16bf f;
  __builtin_memcpy(reinterpret_cast<char*>(&f),      rowp + kbase +     hi*16, 16);
  __builtin_memcpy(reinterpret_cast<char*>(&f) + 16, rowp + kbase + 8 + hi*16, 16);
  return f;
}

// B fragment from pre-packed bf16 weights: lane's 16 elements contiguous (32B),
// lanes consecutive -> two global_load_b128, fully coalesced, zero conversion VALU.
__device__ inline v16bf load_b_frag_packed(const unsigned short* __restrict__ p){
  v16bf f;
  __builtin_memcpy(&f, p, 32);
  return f;
}

// ---------------- x -> bf16 copy (once; reused by all up-proj GEMMs) ----------------
__global__ __launch_bounds__(256) void xbf_kernel(const float* __restrict__ x,
                                                  unsigned short* __restrict__ xb){
  const size_t d = (size_t)blockIdx.x * 256 + threadIdx.x;   // dword index
  const float2 v = reinterpret_cast<const float2*>(x)[d];
  reinterpret_cast<unsigned*>(xb)[d] = pack2bf(v.x, v.y);
}

// ---------------- weight repack: fp32 [mat][K][N] -> bf16 B-fragment layout ----------------
// out flat index per mat: (((nt*KT + kt)*32 + lane)*16 + j), element j of lane L maps to
// k = kt*32 + (L>>4)*16 + j, n = nt*16 + (L&15).
template<int K, int N>
__global__ __launch_bounds__(256) void repack_kernel(const float* __restrict__ in,
                                                     unsigned short* __restrict__ out){
  constexpr int KT = K / 32;
  const int mat = blockIdx.y;
  const size_t d = (size_t)blockIdx.x * 256 + threadIdx.x;   // dword index within mat
  const size_t f = d * 2;                                    // first bf16 element index
  const int j   = (int)(f & 15);
  const int L   = (int)((f >> 4) & 31);
  const size_t tile = f >> 9;                                // nt*KT + kt
  const int kt = (int)(tile % KT);
  const int nt = (int)(tile / KT);
  const int k0 = kt*32 + (L >> 4)*16 + j;
  const int n  = nt*16 + (L & 15);
  const float* src = in + (size_t)mat * K * N;
  const float a = src[(size_t)k0 * N + n];
  const float b = src[(size_t)(k0 + 1) * N + n];
  *reinterpret_cast<unsigned*>(out + (size_t)mat * ((size_t)K * N) + f) = pack2bf(a, b);
}

// ---------------- init ----------------
__global__ void zero_out_kernel(float* __restrict__ out){
  const size_t i = (size_t)blockIdx.x * blockDim.x + threadIdx.x;
  float4 z; z.x = z.y = z.z = z.w = 0.f;
  reinterpret_cast<float4*>(out)[i] = z;
}
__global__ void zero_counts_kernel(int* __restrict__ counts){
  if (threadIdx.x < kE) counts[threadIdx.x] = 0;
}

// ---------------- routing: one wave per token, lane == expert ----------------
__global__ __launch_bounds__(256) void routing_kernel(
    const float* __restrict__ x, const float* __restrict__ gate_w,
    int* __restrict__ counts, int* __restrict__ tok_list,
    float* __restrict__ tok_w, float* __restrict__ cw_small)
{
  const int lane = threadIdx.x & 31;
  const int wv   = threadIdx.x >> 5;
  const int t    = blockIdx.x * 8 + wv;

  const float* xr = x + (size_t)t * kH;
  const float* gr = gate_w + (size_t)lane * kH;
  float acc = 0.f;
  for (int h = 0; h < kH; h += 4) {
    float4 xv = *reinterpret_cast<const float4*>(xr + h);
    float4 gv = *reinterpret_cast<const float4*>(gr + h);
    acc += xv.x*gv.x + xv.y*gv.y + xv.z*gv.z + xv.w*gv.w;
  }
  const float logit = acc;

  float mx = logit;
#pragma unroll
  for (int o = 16; o; o >>= 1) mx = fmaxf(mx, __shfl_xor(mx, o));
  const float ex = __expf(logit - mx);
  float s = ex;
#pragma unroll
  for (int o = 16; o; o >>= 1) s += __shfl_xor(s, o);
  const float rw = ex / s;

  float sc = sigmoidf_(logit);          // top-4 by sigmoid score, low lane wins ties
  int   sel[kTOPK];
  float w[kTOPK];
#pragma unroll
  for (int i = 0; i < kTOPK; ++i) {
    float m = sc; int idx = lane;
#pragma unroll
    for (int o = 16; o; o >>= 1) {
      float om = __shfl_xor(m, o);
      int   oi = __shfl_xor(idx, o);
      if (om > m || (om == m && oi < idx)) { m = om; idx = oi; }
    }
    sel[i] = idx;
    w[i]   = __shfl(rw, idx);
    if (lane == idx) sc = -1.f;
  }
  float ws = w[0] + w[1] + w[2] + w[3];
#pragma unroll
  for (int i = 0; i < kTOPK; ++i) w[i] /= ws;

  if (lane == 0) {
#pragma unroll
    for (int i = 0; i < kTOPK; ++i) {
      const int e = sel[i];
      const int p = atomicAdd(&counts[e], 1);
      tok_list[e * kT + p] = t;
      tok_w[e * kT + p]    = w[i];
    }
  }
  if (lane < kNC) {
    float cw = 0.f;
#pragma unroll
    for (int i = 0; i < kTOPK; ++i) if ((sel[i] >> 1) == lane) cw += w[i];
    cw_small[t * kNC + lane] = cw;
  }
}

__global__ void prefix_kernel(const int* __restrict__ counts, int* __restrict__ offsets){
  if (threadIdx.x == 0) {
    int s = 0;
    for (int e = 0; e < kE; ++e) { offsets[e] = s; s += counts[e]; }
    offsets[kE] = s;
  }
}

// ---------------- large experts: gathered up-proj (g,u) ----------------
// block: 256 thr (8 waves); tile = 16 gathered tokens x full I (48 n-tiles, 6/wave).
// Software-pipelined: next B (and next A at the kt boundary) preloaded into separate
// registers before the WMMAs consuming the current fragments.
__global__ __launch_bounds__(256) void up_large_kernel(
    const unsigned short* __restrict__ xb,
    const unsigned short* __restrict__ WgP, const unsigned short* __restrict__ WuP,
    const int* __restrict__ counts, const int* __restrict__ offsets,
    const int* __restrict__ tok_list, const float* __restrict__ tok_w,
    unsigned short* __restrict__ a_large)
{
  constexpr int KT = kH / 32;   // 64
  const int e = blockIdx.y;
  const int cnt = counts[e];
  const int rbase = blockIdx.x * 16;
  if (rbase >= cnt) return;

  __shared__ int   tl[16];
  __shared__ float wl[16];
  const int tid = threadIdx.x, lane = tid & 31, wv = tid >> 5, hi = lane >> 4, n = lane & 15;

  if (tid < 16) {
    const int r = rbase + tid;
    const bool v = (r < cnt);
    tl[tid] = v ? tok_list[e * kT + r] : 0;
    wl[tid] = v ? tok_w[e * kT + r] : 0.f;
  }
  __syncthreads();

  const v8f vz = {0.f,0.f,0.f,0.f,0.f,0.f,0.f,0.f};
  v8f gacc[6], uacc[6];
#pragma unroll
  for (int i = 0; i < 6; ++i) { gacc[i] = vz; uacc[i] = vz; }

  const unsigned short* aRow = xb + (size_t)tl[n] * kH;   // lane's A row (m = lane&15)
  const unsigned short* wgB = WgP + (size_t)e * ((size_t)kH * kI) + (size_t)(wv*6) * KT * 512 + lane*16;
  const unsigned short* wuB = WuP + (size_t)e * ((size_t)kH * kI) + (size_t)(wv*6) * KT * 512 + lane*16;

  v16bf aF  = load_a_frag_g(aRow, 0, hi);
  v16bf aFn = aF;
  v16bf bg  = load_b_frag_packed(wgB);
  v16bf bu  = load_b_frag_packed(wuB);

  for (int kt = 0; kt < KT - 1; ++kt) {
#pragma unroll
    for (int nt = 0; nt < 6; ++nt) {
      const size_t noff = (nt < 5) ? ((size_t)(nt + 1) * KT + kt) * 512
                                   : (size_t)(kt + 1) * 512;
      const v16bf bgn = load_b_frag_packed(wgB + noff);
      const v16bf bun = load_b_frag_packed(wuB + noff);
      if (nt == 5) aFn = load_a_frag_g(aRow, (kt + 1) * 32, hi);
      gacc[nt] = wmma_bf16(aF, bg, gacc[nt]);
      uacc[nt] = wmma_bf16(aF, bu, uacc[nt]);
      bg = bgn; bu = bun;
    }
    aF = aFn;
  }
  { // tail: kt = KT-1
    const int kt = KT - 1;
#pragma unroll
    for (int nt = 0; nt < 6; ++nt) {
      v16bf bgn = bg, bun = bu;
      if (nt < 5) {
        const size_t noff = ((size_t)(nt + 1) * KT + kt) * 512;
        bgn = load_b_frag_packed(wgB + noff);
        bun = load_b_frag_packed(wuB + noff);
      }
      gacc[nt] = wmma_bf16(aF, bg, gacc[nt]);
      uacc[nt] = wmma_bf16(aF, bu, uacc[nt]);
      bg = bgn; bu = bun;
    }
  }

  const int obase = offsets[e] + rbase;
#pragma unroll
  for (int nt = 0; nt < 6; ++nt) {
    const int col = wv * 96 + nt * 16 + n;
#pragma unroll
    for (int v = 0; v < 8; ++v) {
      const int m = v + 8 * hi;
      if (rbase + m < cnt) {
        const float a = siluf_(gacc[nt][v]) * uacc[nt][v] * wl[m];
        a_large[(size_t)(obase + m) * kI + col] = f2bfbits(a);
      }
    }
  }
}

// ---------------- large experts: gathered down-proj + scatter-add ----------------
__global__ __launch_bounds__(256) void down_large_kernel(
    const unsigned short* __restrict__ a_large, const unsigned short* __restrict__ WdP,
    const int* __restrict__ counts, const int* __restrict__ offsets,
    const int* __restrict__ tok_list, float* __restrict__ out)
{
  constexpr int KT = kI / 32;   // 24
  constexpr int NT = kH / 16;   // 128
  const int e = blockIdx.y;
  const int cnt = counts[e];
  const int rbase = blockIdx.x * 16;
  if (rbase >= cnt) return;

  const int tid = threadIdx.x, lane = tid & 31, wv = tid >> 5, hi = lane >> 4, n = lane & 15;
  const int obase = offsets[e] + rbase;
  const int nt = blockIdx.z * 8 + wv;                 // 8 n-tiles per block, one per wave
  const unsigned short* wdB = WdP + ((size_t)e * NT + nt) * ((size_t)KT * 512) + lane*16;

  // lane's A row; clamp overshoot rows into the buffer (results discarded in epilogue)
  int rowm = obase + n; if (rowm > kT * kTOPK - 1) rowm = kT * kTOPK - 1;
  const unsigned short* aRow = a_large + (size_t)rowm * kI;

  v8f acc = {0.f,0.f,0.f,0.f,0.f,0.f,0.f,0.f};
  v16bf aF = load_a_frag_g(aRow, 0, hi);
  v16bf bF = load_b_frag_packed(wdB);
  for (int kt = 0; kt < KT - 1; ++kt) {
    const v16bf aFn = load_a_frag_g(aRow, (kt + 1) * 32, hi);
    const v16bf bFn = load_b_frag_packed(wdB + (size_t)(kt + 1) * 512);
    acc = wmma_bf16(aF, bF, acc);
    aF = aFn; bF = bFn;
  }
  acc = wmma_bf16(aF, bF, acc);

#pragma unroll
  for (int v = 0; v < 8; ++v) {
    const int m = v + 8 * hi;
    if (rbase + m < cnt) {
      const int t = tok_list[e * kT + rbase + m];
      atomicAdd(&out[(size_t)t * kH + nt*16 + n], acc[v]);
    }
  }
}

// ---------------- small (chunk) experts: dense up-proj ----------------
__global__ __launch_bounds__(256) void up_small_kernel(
    const unsigned short* __restrict__ xb,
    const unsigned short* __restrict__ cWgP, const unsigned short* __restrict__ cWuP,
    const float* __restrict__ cw_small, unsigned short* __restrict__ a_small)
{
  constexpr int KT = kH / 32;   // 64
  const int c  = blockIdx.y;
  const int tb = blockIdx.x * 16;
  const int tid = threadIdx.x, lane = tid & 31, wv = tid >> 5, hi = lane >> 4, n = lane & 15;

  v8f gacc = {0.f,0.f,0.f,0.f,0.f,0.f,0.f,0.f};
  v8f uacc = {0.f,0.f,0.f,0.f,0.f,0.f,0.f,0.f};
  const unsigned short* aRow = xb + (size_t)(tb + n) * kH;
  const unsigned short* wgB = cWgP + ((size_t)c * 8 + wv) * ((size_t)KT * 512) + lane*16;  // nt == wv
  const unsigned short* wuB = cWuP + ((size_t)c * 8 + wv) * ((size_t)KT * 512) + lane*16;

  v16bf aF = load_a_frag_g(aRow, 0, hi);
  v16bf bg = load_b_frag_packed(wgB);
  v16bf bu = load_b_frag_packed(wuB);
  for (int kt = 0; kt < KT - 1; ++kt) {
    const v16bf aFn = load_a_frag_g(aRow, (kt + 1) * 32, hi);
    const v16bf bgn = load_b_frag_packed(wgB + (size_t)(kt + 1) * 512);
    const v16bf bun = load_b_frag_packed(wuB + (size_t)(kt + 1) * 512);
    gacc = wmma_bf16(aF, bg, gacc);
    uacc = wmma_bf16(aF, bu, uacc);
    aF = aFn; bg = bgn; bu = bun;
  }
  gacc = wmma_bf16(aF, bg, gacc);
  uacc = wmma_bf16(aF, bu, uacc);

#pragma unroll
  for (int v = 0; v < 8; ++v) {
    const int m = v + 8 * hi;
    const int t = tb + m;
    const float cw = cw_small[t * kNC + c];
    const float a = siluf_(gacc[v]) * uacc[v] * cw * kSMALL;   // 0.05 folded in here
    a_small[((size_t)c * kT + t) * kCI + wv*16 + n] = f2bfbits(a);
  }
}

// ---------------- small experts: dense down-proj + scatter-add ----------------
__global__ __launch_bounds__(256) void down_small_kernel(
    const unsigned short* __restrict__ a_small, const unsigned short* __restrict__ cWdP,
    float* __restrict__ out)
{
  constexpr int KT = kCI / 32;  // 4
  constexpr int NT = kH / 16;   // 128
  const int c  = blockIdx.y;
  const int tb = blockIdx.x * 16;
  const int tid = threadIdx.x, lane = tid & 31, wv = tid >> 5, hi = lane >> 4, n = lane & 15;
  const int nt = blockIdx.z * 8 + wv;
  const unsigned short* wdB = cWdP + ((size_t)c * NT + nt) * ((size_t)KT * 512) + lane*16;
  const unsigned short* aRow = a_small + ((size_t)c * kT + tb + n) * kCI;

  v8f acc = {0.f,0.f,0.f,0.f,0.f,0.f,0.f,0.f};
  v16bf aF = load_a_frag_g(aRow, 0, hi);
  v16bf bF = load_b_frag_packed(wdB);
#pragma unroll
  for (int kt = 0; kt < KT - 1; ++kt) {
    const v16bf aFn = load_a_frag_g(aRow, (kt + 1) * 32, hi);
    const v16bf bFn = load_b_frag_packed(wdB + (size_t)(kt + 1) * 512);
    acc = wmma_bf16(aF, bF, acc);
    aF = aFn; bF = bFn;
  }
  acc = wmma_bf16(aF, bF, acc);

#pragma unroll
  for (int v = 0; v < 8; ++v) {
    const int m = v + 8 * hi;
    atomicAdd(&out[(size_t)(tb + m) * kH + nt*16 + n], acc[v]);
  }
}

// ---------------- launch ----------------
extern "C" void kernel_launch(void* const* d_in, const int* in_sizes, int n_in,
                              void* d_out, int out_size, void* d_ws, size_t ws_size,
                              hipStream_t stream)
{
  const float* x      = (const float*)d_in[0];
  const float* gate_w = (const float*)d_in[1];
  const float* Wg     = (const float*)d_in[2];
  const float* Wu     = (const float*)d_in[3];
  const float* Wd     = (const float*)d_in[4];
  const float* cWg    = (const float*)d_in[5];
  const float* cWu    = (const float*)d_in[6];
  const float* cWd    = (const float*)d_in[7];
  float* out = (float*)d_out;

  // workspace layout (~388 MB: routing/activations + bf16 x + packed bf16 weights)
  char* ws = (char*)d_ws;
  size_t off = 0;
  auto alloc = [&](size_t bytes) -> void* {
    void* p = ws + off;
    off += (bytes + 255) & ~(size_t)255;
    return p;
  };
  int*            counts   = (int*)           alloc((size_t)kE * 4);
  int*            offsets  = (int*)           alloc((size_t)(kE + 1) * 4);
  int*            tok_list = (int*)           alloc((size_t)kE * kT * 4);
  float*          tok_w    = (float*)         alloc((size_t)kE * kT * 4);
  float*          cw_small = (float*)         alloc((size_t)kT * kNC * 4);
  unsigned short* xb       = (unsigned short*)alloc((size_t)kT * kH * 2);
  unsigned short* a_large  = (unsigned short*)alloc(((size_t)kT * kTOPK + 16) * kI * 2); // +16 rows clamp slack
  unsigned short* a_small  = (unsigned short*)alloc((size_t)kNC * kT * kCI * 2);
  unsigned short* WgP      = (unsigned short*)alloc((size_t)kE * kH * kI * 2);
  unsigned short* WuP      = (unsigned short*)alloc((size_t)kE * kH * kI * 2);
  unsigned short* WdP      = (unsigned short*)alloc((size_t)kE * kI * kH * 2);
  unsigned short* cWgP     = (unsigned short*)alloc((size_t)kNC * kH * kCI * 2);
  unsigned short* cWuP     = (unsigned short*)alloc((size_t)kNC * kH * kCI * 2);
  unsigned short* cWdP     = (unsigned short*)alloc((size_t)kNC * kCI * kH * 2);
  (void)ws_size; (void)in_sizes; (void)n_in; (void)out_size;

  // one-time conversions / repacks (bandwidth-bound, ~45 us total at 23.3 TB/s)
  xbf_kernel<<<(kT * kH / 2) / 256, 256, 0, stream>>>(x, xb);
  repack_kernel<kH, kI><<<dim3(kH*kI/512, kE), 256, 0, stream>>>(Wg,  WgP);
  repack_kernel<kH, kI><<<dim3(kH*kI/512, kE), 256, 0, stream>>>(Wu,  WuP);
  repack_kernel<kI, kH><<<dim3(kI*kH/512, kE), 256, 0, stream>>>(Wd,  WdP);
  repack_kernel<kH, kCI><<<dim3(kH*kCI/512, kNC), 256, 0, stream>>>(cWg, cWgP);
  repack_kernel<kH, kCI><<<dim3(kH*kCI/512, kNC), 256, 0, stream>>>(cWu, cWuP);
  repack_kernel<kCI, kH><<<dim3(kCI*kH/512, kNC), 256, 0, stream>>>(cWd, cWdP);

  zero_out_kernel   <<<(kT * kH / 4) / 256, 256, 0, stream>>>(out);
  zero_counts_kernel<<<1, 32, 0, stream>>>(counts);
  routing_kernel    <<<kT / 8, 256, 0, stream>>>(x, gate_w, counts, tok_list, tok_w, cw_small);
  prefix_kernel     <<<1, 32, 0, stream>>>(counts, offsets);

  up_large_kernel   <<<dim3(kT / 16, kE, 1),        256, 0, stream>>>(xb, WgP, WuP, counts, offsets, tok_list, tok_w, a_large);
  down_large_kernel <<<dim3(kT / 16, kE, kH / 128), 256, 0, stream>>>(a_large, WdP, counts, offsets, tok_list, out);
  up_small_kernel   <<<dim3(kT / 16, kNC, 1),        256, 0, stream>>>(xb, cWgP, cWuP, cw_small, a_small);
  down_small_kernel <<<dim3(kT / 16, kNC, kH / 128), 256, 0, stream>>>(a_small, cWdP, out);
}
